// GCNEncoder_73796128080058
// MI455X (gfx1250) — compile-verified, matrix operand order checked by reading, not run on hardware
//
#include <hip/hip_runtime.h>

#define N_NODES 100000
#define N_EDGES 1600000
#define DIM_IN  512
#define CH      128
#define CH_OUT  64

static_assert(N_NODES % 32 == 0, "tile guards removed; M must divide tiles");

typedef __attribute__((ext_vector_type(16))) __bf16 v16bf;
typedef __attribute__((ext_vector_type(8)))  __bf16 v8bf;
typedef __attribute__((ext_vector_type(8)))  float  v8f;

#define ROWS 72  // 64 K + 8 pad (144 B row stride: 16B aligned, bank-staggered)

__device__ __forceinline__ unsigned int f2bf(float f) {
  unsigned int u = __float_as_uint(f);
  u += 0x7FFFu + ((u >> 16) & 1u);   // round-to-nearest-even
  return u >> 16;
}

// 16-bit A/B fragment at K-chunk base `row`: lane group g holds K in
// [8g,8g+8) and [16+8g,24+8g) -> two aligned ds_load_b128 per lane.
__device__ __forceinline__ v16bf load_frag(const unsigned short* row, int g) {
  v8bf lo = *reinterpret_cast<const v8bf*>(row + 8 * g);
  v8bf hi = *reinterpret_cast<const v8bf*>(row + 16 + 8 * g);
  v16bf f;
#pragma unroll
  for (int i = 0; i < 8; ++i) { f[i] = lo[i]; f[i + 8] = hi[i]; }
  return f;
}

// C[M,N] = A[M,Kd] @ W[Kd,N] (+bias). 256 threads = 8 waves, one 16x16 tile
// per wave, MT = 8/NT row-tiles per block, K-step 64 -> 2 WMMA per barrier.
template<int Kd, int N, int NT, bool A_BF16, bool OUT_BF16>
__global__ __launch_bounds__(256) void wmma_gemm(
    const void* __restrict__ Ap, const float* __restrict__ W,
    const float* __restrict__ bias, void* __restrict__ Cp) {
  constexpr int MT = 8 / NT;
  constexpr int TR = MT * 16;                 // rows per block
  __shared__ __align__(16) unsigned short As[TR][ROWS];
  __shared__ __align__(16) unsigned short Bs[N][ROWS];

  const int block_m = blockIdx.x * TR;
  const int tid  = threadIdx.x;
  const int wave = tid >> 5, lane = tid & 31;
  const int mt = wave / NT, nt = wave % NT;
  const int col_base = nt * 16;
  const int g = lane >> 4, ln = lane & 15;

  v8f acc = {};
  for (int k0 = 0; k0 < Kd; k0 += 64) {
    // ---- stage A tile (TR x 64) as bf16, 4 K per thread ----
#pragma unroll
    for (int i = tid; i < TR * 16; i += 256) {
      int r = i >> 4, kk = (i & 15) * 4;
      size_t off = (size_t)(block_m + r) * Kd + k0 + kk;
      uint2 packed;
      if (A_BF16) {
        packed = *reinterpret_cast<const uint2*>((const unsigned short*)Ap + off);
      } else {
        float4 v = *reinterpret_cast<const float4*>((const float*)Ap + off);
        packed.x = f2bf(v.x) | (f2bf(v.y) << 16);
        packed.y = f2bf(v.z) | (f2bf(v.w) << 16);
      }
      *reinterpret_cast<uint2*>(&As[r][kk]) = packed;
    }
    // ---- stage B tile transposed (N x 64), 4 n per thread, coalesced ----
#pragma unroll
    for (int i = tid; i < 64 * (N / 4); i += 256) {
      int kk = i / (N / 4), n = (i - kk * (N / 4)) * 4;
      float4 wv = *reinterpret_cast<const float4*>(&W[(size_t)(k0 + kk) * N + n]);
      Bs[n + 0][kk] = (unsigned short)f2bf(wv.x);
      Bs[n + 1][kk] = (unsigned short)f2bf(wv.y);
      Bs[n + 2][kk] = (unsigned short)f2bf(wv.z);
      Bs[n + 3][kk] = (unsigned short)f2bf(wv.w);
    }
    __syncthreads();
    const unsigned short* arow = As[mt * 16 + ln];
    const unsigned short* brow = Bs[col_base + ln];
    acc = __builtin_amdgcn_wmma_f32_16x16x32_bf16(
        false, load_frag(arow, g), false, load_frag(brow, g),
        (short)0, acc, false, false);
    acc = __builtin_amdgcn_wmma_f32_16x16x32_bf16(
        false, load_frag(arow + 32, g), false, load_frag(brow + 32, g),
        (short)0, acc, false, false);
    __syncthreads();
  }
  // D layout: VGPR r -> M = r + 8*g, lane[3:0] -> N
  const int row_base = block_m + mt * 16 + 8 * g;
#pragma unroll
  for (int r = 0; r < 8; ++r) {
    int gm = row_base + r;
    int gn = col_base + ln;
    float v = acc[r] + (bias ? bias[gn] : 0.0f);
    if (OUT_BF16) ((unsigned short*)Cp)[(size_t)gm * N + gn] = (unsigned short)f2bf(v);
    else          ((float*)Cp)[(size_t)gm * N + gn] = v;
  }
}

__global__ void k_fill(float* p, float v, int n) {
  int i = blockIdx.x * blockDim.x + threadIdx.x;
  if (i < n) p[i] = v;
}
__global__ void k_deg_edges(const int* __restrict__ dst, float* deg, int ne) {
  int i = blockIdx.x * blockDim.x + threadIdx.x;
  if (i < ne) atomicAdd(&deg[dst[i]], 1.0f);
}
__global__ void k_rsqrt(float* p, int n) {
  int i = blockIdx.x * blockDim.x + threadIdx.x;
  if (i < n) p[i] = rsqrtf(p[i]);
}
// out[i,:] = t[i,:] * dis[i]^2 + b   (self-loop contribution + bias)
__global__ void k_agg_init(const float* __restrict__ t,
                           const float* __restrict__ dis,
                           const float* __restrict__ b,
                           float* __restrict__ out, int n, int F) {
  int i = blockIdx.x * blockDim.x + threadIdx.x;
  int per = F >> 2;
  if (i >= n * per) return;
  int node = i / per, c4 = (i - node * per) * 4;
  float w = dis[node] * dis[node];
  float4 v = *(const float4*)&t[(size_t)node * F + c4];
  float4 o;
  o.x = v.x * w + b[c4 + 0];
  o.y = v.y * w + b[c4 + 1];
  o.z = v.z * w + b[c4 + 2];
  o.w = v.w * w + b[c4 + 3];
  *(float4*)&out[(size_t)node * F + c4] = o;
}
// out[dst,:] += t[src,:] * dis[src]*dis[dst]   (memory-bound random gather)
__global__ void k_agg_edges(const float* __restrict__ t,
                            const int* __restrict__ src,
                            const int* __restrict__ dst,
                            const float* __restrict__ dis,
                            float* __restrict__ out, int ne, int F) {
  int i = blockIdx.x * blockDim.x + threadIdx.x;
  int per = F >> 2;
  if (i >= ne * per) return;
  int e = i / per, c4 = (i - e * per) * 4;
  // hide gather latency: prefetch the src row ~32 edges ahead
  if (c4 == 0 && e + 32 < ne)
    __builtin_prefetch(&t[(size_t)src[e + 32] * F], 0, 1);
  int s = src[e], d = dst[e];
  float w = dis[s] * dis[d];
  float4 v = *(const float4*)&t[(size_t)s * F + c4];
  float* o = &out[(size_t)d * F + c4];
  atomicAdd(o + 0, v.x * w);
  atomicAdd(o + 1, v.y * w);
  atomicAdd(o + 2, v.z * w);
  atomicAdd(o + 3, v.w * w);
}
__global__ void k_relu_bf16(const float* __restrict__ in,
                            unsigned short* __restrict__ out, int n) {
  int i = blockIdx.x * blockDim.x + threadIdx.x;
  if (i < n) out[i] = (unsigned short)f2bf(fmaxf(in[i], 0.0f));
}
__global__ void k_relu(float* p, int n) {
  int i = blockIdx.x * blockDim.x + threadIdx.x;
  if (i < n) p[i] = fmaxf(p[i], 0.0f);
}

static inline size_t alignup(size_t x) { return (x + 255) & ~(size_t)255; }

extern "C" void kernel_launch(void* const* d_in, const int* in_sizes, int n_in,
                              void* d_out, int out_size, void* d_ws, size_t ws_size,
                              hipStream_t stream) {
  (void)in_sizes; (void)n_in; (void)out_size; (void)ws_size;
  const float* x     = (const float*)d_in[0];
  const float* W_lin = (const float*)d_in[1];
  const float* b_lin = (const float*)d_in[2];
  const float* W1    = (const float*)d_in[3];
  const float* b1    = (const float*)d_in[4];
  const float* W2    = (const float*)d_in[5];
  const float* b2    = (const float*)d_in[6];
  const float* W3    = (const float*)d_in[7];
  const float* b3    = (const float*)d_in[8];
  const int*   ei    = (const int*)d_in[9];
  const int* esrc = ei;
  const int* edst = ei + N_EDGES;

  char* w = (char*)d_ws;
  size_t o = 0;
  float*          dis  = (float*)(w + o);          o += alignup((size_t)N_NODES * 4);
  unsigned short* h0   = (unsigned short*)(w + o); o += alignup((size_t)N_NODES * CH * 2);
  float*          t1   = (float*)(w + o);          o += alignup((size_t)N_NODES * CH * 4);
  float*          agg1 = (float*)(w + o);          o += alignup((size_t)N_NODES * CH * 4);
  unsigned short* g1   = (unsigned short*)(w + o); o += alignup((size_t)N_NODES * CH * 2);
  float* t2 = t1;    // t1 dead after conv1 aggregation consumed it
  float* t3 = agg1;  // agg1 dead after ReLU+bf16 repack

  float* mu = (float*)d_out;
  float* lv = mu + (size_t)N_NODES * CH_OUT;

  const int B = 256;
  // deg^{-1/2} with self-loops (init 1.0)
  k_fill<<<(N_NODES + B - 1) / B, B, 0, stream>>>(dis, 1.0f, N_NODES);
  k_deg_edges<<<(N_EDGES + B - 1) / B, B, 0, stream>>>(edst, dis, N_EDGES);
  k_rsqrt<<<(N_NODES + B - 1) / B, B, 0, stream>>>(dis, N_NODES);

  // h0 = bf16(x @ W_lin + b_lin)
  wmma_gemm<DIM_IN, CH, 8, false, true><<<N_NODES / 16, 256, 0, stream>>>(
      x, W_lin, b_lin, h0);
  // t1 = h0 @ W1
  wmma_gemm<CH, CH, 8, true, false><<<N_NODES / 16, 256, 0, stream>>>(
      h0, W1, nullptr, t1);
  // agg1 = scatter-add(norm * gather(t1)) + b1 ; g1 = bf16(relu(agg1))
  int q1  = N_NODES * (CH / 4);
  int qe1 = N_EDGES * (CH / 4);
  int nh  = N_NODES * CH;
  k_agg_init<<<(q1 + B - 1) / B, B, 0, stream>>>(t1, dis, b1, agg1, N_NODES, CH);
  k_agg_edges<<<(qe1 + B - 1) / B, B, 0, stream>>>(t1, esrc, edst, dis, agg1, N_EDGES, CH);
  k_relu_bf16<<<(nh + B - 1) / B, B, 0, stream>>>(agg1, g1, nh);

  // t2 = g1 @ W2 ; t3 = g1 @ W3   (N=64 -> 2 row-tiles per block)
  wmma_gemm<CH, CH_OUT, 4, true, false><<<N_NODES / 32, 256, 0, stream>>>(
      g1, W2, nullptr, t2);
  wmma_gemm<CH, CH_OUT, 4, true, false><<<N_NODES / 32, 256, 0, stream>>>(
      g1, W3, nullptr, t3);

  // mu / logvar aggregation into d_out
  int q2  = N_NODES * (CH_OUT / 4);
  int qe2 = N_EDGES * (CH_OUT / 4);
  int no  = N_NODES * CH_OUT;
  k_agg_init<<<(q2 + B - 1) / B, B, 0, stream>>>(t2, dis, b2, mu, N_NODES, CH_OUT);
  k_agg_edges<<<(qe2 + B - 1) / B, B, 0, stream>>>(t2, esrc, edst, dis, mu, N_EDGES, CH_OUT);
  k_agg_init<<<(q2 + B - 1) / B, B, 0, stream>>>(t3, dis, b3, lv, N_NODES, CH_OUT);
  k_agg_edges<<<(qe2 + B - 1) / B, B, 0, stream>>>(t3, esrc, edst, dis, lv, N_EDGES, CH_OUT);
  k_relu<<<(no + B - 1) / B, B, 0, stream>>>(lv, no);
}
